// ShahCapSAREncoder_64141041599138
// MI455X (gfx1250) — compile-verified, matrix operand order checked by reading, not run on hardware
//
#include <hip/hip_runtime.h>

#if defined(__has_builtin)
#  if __has_builtin(__builtin_amdgcn_global_load_async_to_lds_b128)
#    define HAVE_ASYNC_LDS 1
#  endif
#endif
#ifndef HAVE_ASYNC_LDS
#  define HAVE_ASYNC_LDS 0
#endif

typedef __attribute__((ext_vector_type(16))) _Float16 v16h;
typedef __attribute__((ext_vector_type(8)))  _Float16 v8h;
typedef __attribute__((ext_vector_type(8)))  float    v8f;

union ABFrag { v16h v; v8h h[2]; };

#if HAVE_ASYNC_LDS
// Builtin parameter type (from hipcc diagnostic): pointer to GCC-style int4
// vector in AS1 (global) / AS3 (LDS).
typedef int gv4i __attribute__((vector_size(16)));
typedef __attribute__((address_space(1))) gv4i* as1_v4i_p;
typedef __attribute__((address_space(3))) gv4i* as3_v4i_p;
#endif

// 16-byte global->LDS copy. Async path (GLOBAL_LOAD_ASYNC_TO_LDS_B128, tracked
// by ASYNCcnt) when available, else plain load + ds_store_b128.
__device__ __forceinline__ void cp16(const _Float16* g, _Float16* l) {
#if HAVE_ASYNC_LDS
    __builtin_amdgcn_global_load_async_to_lds_b128(
        (as1_v4i_p)(unsigned long long)g,
        (as3_v4i_p)(unsigned int)(unsigned long long)l,
        0, 0);
#else
    *(v8h*)l = *(const v8h*)g;
#endif
}
__device__ __forceinline__ void cp_commit() {
#if HAVE_ASYNC_LDS
    asm volatile("s_wait_asynccnt 0" ::: "memory");
#endif
}

// ---------------------------------------------------------------------------
// Weight reformat: [O=256][I=256][taps] f32  ->  [n=256][k=taps*256] f16
// ---------------------------------------------------------------------------
__global__ void refmt_k(const float* __restrict__ W, _Float16* __restrict__ Wt, int taps) {
    int t = blockIdx.x * 256 + threadIdx.x;
    int total = 256 * taps * 256;
    if (t >= total) return;
    int c   = t & 255;
    int r   = t >> 8;
    int nn  = r / taps;
    int tap = r % taps;
    Wt[((size_t)nn * taps + tap) * 256 + c] =
        (_Float16)W[(((size_t)nn * 256) + c) * taps + tap];
}

// ---------------------------------------------------------------------------
// conv1: [4,1,131,131] * [256,1,9,9] -> ReLU -> act1 f16 NHWC [4,123,123,256]
// ---------------------------------------------------------------------------
__global__ void conv1_k(const float* __restrict__ X, const float* __restrict__ W1,
                        const float* __restrict__ B1, _Float16* __restrict__ act1) {
    int t = blockIdx.x * 256 + threadIdx.x;              // 4*15129*256 threads
    int c = t & 255;
    int p = t >> 8;                                      // b*15129 + pix
    int b = p / 15129;
    int pix = p % 15129;
    int y = pix / 123, xx = pix % 123;
    const float* xp = X + (size_t)(b * 131 + y) * 131 + xx;
    const float* wp = W1 + c * 81;
    float acc = B1[c];
#pragma unroll
    for (int ky = 0; ky < 9; ++ky)
#pragma unroll
        for (int kx = 0; kx < 9; ++kx)
            acc += xp[ky * 131 + kx] * wp[ky * 9 + kx];
    acc = acc > 0.f ? acc : 0.f;
    act1[(size_t)p * 256 + c] = (_Float16)acc;
}

// ---------------------------------------------------------------------------
// conv2 implicit GEMM, LDS double-buffered.
// Block tile 64M x 256N (8 waves = 2x4, wave tile 32x64, 8 wmma / K-step).
// K = 25 taps * 256 ch = 6400 -> 200 steps of 32.
// LDS rows padded to 40 halves (80B) for bank spread; 16B aligned.
// ---------------------------------------------------------------------------
__global__ __launch_bounds__(256) void conv2_wmma(
    const _Float16* __restrict__ act1, const _Float16* __restrict__ w,
    const float* __restrict__ bias, _Float16* __restrict__ act2) {
    __shared__ alignas(16) _Float16 sm[2][12800];        // 2 x (64+256)*40 f16 = 50 KB
    const int t    = threadIdx.x;
    const int lane = t & 31, wave = t >> 5;
    const int l16  = lane & 15, hf = lane >> 4;
    const int mw   = wave >> 2;                          // 0..1 : 32-row sub-block
    const int nw   = wave & 3;                           // 0..3 : 64-col sub-block
    const int mb   = blockIdx.x;                         // 0..221
    const int b    = blockIdx.y;

    // staging roles: A -> thread = (row 0..63, kq 0..3); B -> thread = col n
    const int arow = t >> 2, akq = t & 3;
    int apix = mb * 64 + arow;
    if (apix > 14160) apix = 14160;                      // clamp, EXEC stays full
    const int aoy = apix / 119, aox = apix % 119;
    const _Float16* aAddr = act1 + ((size_t)(b * 123 + aoy) * 123 + aox) * 256 + akq * 8;
    const _Float16* bAddr = w + (size_t)t * 6400;

    const v8f vzero = {};
    v8f acc[2][4];
#pragma unroll
    for (int s = 0; s < 2; ++s)
#pragma unroll
        for (int u2 = 0; u2 < 4; ++u2) acc[s][u2] = vzero;

    auto stage = [&](int step, int buf) {
        const int tap = step >> 3, c0 = (step & 7) << 5;
        const int ky = tap / 5, kx = tap % 5;
        cp16(aAddr + ((size_t)(ky * 123 + kx)) * 256 + c0, &sm[buf][arow * 40 + akq * 8]);
        const _Float16* gb = bAddr + tap * 256 + c0;
        _Float16* lb = &sm[buf][2560 + t * 40];
        cp16(gb,      lb);
        cp16(gb + 8,  lb + 8);
        cp16(gb + 16, lb + 16);
        cp16(gb + 24, lb + 24);
    };

    stage(0, 0);
    cp_commit();
    __syncthreads();

    for (int step = 0; step < 200; ++step) {
        const int cur = step & 1;
        const _Float16* As = sm[cur];
        const _Float16* Bs = sm[cur] + 2560;
        ABFrag afr[2], bfr[4];
#pragma unroll
        for (int s = 0; s < 2; ++s) {
            const int row = mw * 32 + s * 16 + l16;
            afr[s].h[0] = *(const v8h*)(As + row * 40 + hf * 8);
            afr[s].h[1] = *(const v8h*)(As + row * 40 + hf * 8 + 16);
        }
#pragma unroll
        for (int u2 = 0; u2 < 4; ++u2) {
            const int col = nw * 64 + u2 * 16 + l16;
            bfr[u2].h[0] = *(const v8h*)(Bs + col * 40 + hf * 8);
            bfr[u2].h[1] = *(const v8h*)(Bs + col * 40 + hf * 8 + 16);
        }
        if (step + 1 < 200) stage(step + 1, cur ^ 1);    // overlap with wmma below
#pragma unroll
        for (int s = 0; s < 2; ++s)
#pragma unroll
            for (int u2 = 0; u2 < 4; ++u2)
                acc[s][u2] = __builtin_amdgcn_wmma_f32_16x16x32_f16(
                    false, afr[s].v, false, bfr[u2].v, (short)0, acc[s][u2], false, false);
        cp_commit();
        __syncthreads();
    }

#pragma unroll
    for (int u2 = 0; u2 < 4; ++u2) {
        const int ch = nw * 64 + u2 * 16 + l16;
        const float bs = bias[ch];
#pragma unroll
        for (int s = 0; s < 2; ++s)
#pragma unroll
            for (int r = 0; r < 8; ++r) {
                const int m = mb * 64 + mw * 32 + s * 16 + r + hf * 8;
                if (m < 14161) {
                    float o = acc[s][u2][r] + bs;
                    o = o > 0.f ? o : 0.f;
                    act2[((size_t)b * 14161 + m) * 256 + ch] = (_Float16)o;
                }
            }
    }
}

// ---------------------------------------------------------------------------
// primary-caps conv (9x9, stride 2), same scheme. M=3136/batch (49 blocks of 64),
// K = 81*256 = 20736 -> 648 steps. Output f32 in capsule layout
// u[b][n][i]: n = (c&31)*3136 + pix, i = c>>5.
// ---------------------------------------------------------------------------
__global__ __launch_bounds__(256) void convp_wmma(
    const _Float16* __restrict__ act2, const _Float16* __restrict__ w,
    const float* __restrict__ bias, float* __restrict__ u) {
    __shared__ alignas(16) _Float16 sm[2][12800];
    const int t    = threadIdx.x;
    const int lane = t & 31, wave = t >> 5;
    const int l16  = lane & 15, hf = lane >> 4;
    const int mw   = wave >> 2;
    const int nw   = wave & 3;
    const int mb   = blockIdx.x;                         // 0..48 (49*64 == 3136)
    const int b    = blockIdx.y;

    const int arow = t >> 2, akq = t & 3;
    const int apix = mb * 64 + arow;
    const int aoy = apix / 56, aox = apix % 56;
    const _Float16* aAddr =
        act2 + ((size_t)(b * 14161 + (2 * aoy) * 119 + 2 * aox)) * 256 + akq * 8;
    const _Float16* bAddr = w + (size_t)t * 20736;

    const v8f vzero = {};
    v8f acc[2][4];
#pragma unroll
    for (int s = 0; s < 2; ++s)
#pragma unroll
        for (int u2 = 0; u2 < 4; ++u2) acc[s][u2] = vzero;

    auto stage = [&](int step, int buf) {
        const int tap = step >> 3, c0 = (step & 7) << 5;
        const int ky = tap / 9, kx = tap % 9;
        cp16(aAddr + ((size_t)(ky * 119 + kx)) * 256 + c0, &sm[buf][arow * 40 + akq * 8]);
        const _Float16* gb = bAddr + tap * 256 + c0;
        _Float16* lb = &sm[buf][2560 + t * 40];
        cp16(gb,      lb);
        cp16(gb + 8,  lb + 8);
        cp16(gb + 16, lb + 16);
        cp16(gb + 24, lb + 24);
    };

    stage(0, 0);
    cp_commit();
    __syncthreads();

    for (int step = 0; step < 648; ++step) {
        const int cur = step & 1;
        const _Float16* As = sm[cur];
        const _Float16* Bs = sm[cur] + 2560;
        ABFrag afr[2], bfr[4];
#pragma unroll
        for (int s = 0; s < 2; ++s) {
            const int row = mw * 32 + s * 16 + l16;
            afr[s].h[0] = *(const v8h*)(As + row * 40 + hf * 8);
            afr[s].h[1] = *(const v8h*)(As + row * 40 + hf * 8 + 16);
        }
#pragma unroll
        for (int u2 = 0; u2 < 4; ++u2) {
            const int col = nw * 64 + u2 * 16 + l16;
            bfr[u2].h[0] = *(const v8h*)(Bs + col * 40 + hf * 8);
            bfr[u2].h[1] = *(const v8h*)(Bs + col * 40 + hf * 8 + 16);
        }
        if (step + 1 < 648) stage(step + 1, cur ^ 1);
#pragma unroll
        for (int s = 0; s < 2; ++s)
#pragma unroll
            for (int u2 = 0; u2 < 4; ++u2)
                acc[s][u2] = __builtin_amdgcn_wmma_f32_16x16x32_f16(
                    false, afr[s].v, false, bfr[u2].v, (short)0, acc[s][u2], false, false);
        cp_commit();
        __syncthreads();
    }

#pragma unroll
    for (int u2 = 0; u2 < 4; ++u2) {
        const int ch = nw * 64 + u2 * 16 + l16;
        const float bs = bias[ch];
        const int ci = ch >> 5;
        const int nc = ch & 31;
#pragma unroll
        for (int s = 0; s < 2; ++s)
#pragma unroll
            for (int r = 0; r < 8; ++r) {
                const int m = mb * 64 + mw * 32 + s * 16 + r + hf * 8;
                const int nidx = nc * 3136 + m;
                u[((size_t)b * 100352 + nidx) * 8 + ci] = acc[s][u2][r] + bs;
            }
    }
}

// ---------------------------------------------------------------------------
// squash over i (in place on u)
// ---------------------------------------------------------------------------
__global__ void squash_k(float* __restrict__ u) {
    int t = blockIdx.x * 256 + threadIdx.x;
    if (t >= 401408) return;
    float* p = u + (size_t)t * 8;
    float tmp[8], n2 = 0.f;
#pragma unroll
    for (int i = 0; i < 8; ++i) { tmp[i] = p[i]; n2 += tmp[i] * tmp[i]; }
    float sc = (n2 / (1.f + n2)) * rsqrtf(n2 + 1e-8f);
#pragma unroll
    for (int i = 0; i < 8; ++i) p[i] = tmp[i] * sc;
}

// ---------------------------------------------------------------------------
// u_hat[b][n][j][o] = sum_i u[b][n][i] * Wr[j][n][i][o]; lane = o (coalesced Wr)
// ---------------------------------------------------------------------------
__global__ void uhat_k(const float* __restrict__ u, const float* __restrict__ Wr,
                       float* __restrict__ uhat) {
    int t = blockIdx.x * 256 + threadIdx.x;              // 10*100352*16 threads
    if (t >= 16056320) return;
    int o = t & 15;
    int pair = t >> 4;
    int j = pair / 100352;
    int n = pair % 100352;
    const float* wr = Wr + (((size_t)j * 100352 + n) * 8) * 16 + o;
    float w8[8];
#pragma unroll
    for (int i = 0; i < 8; ++i) w8[i] = wr[i * 16];
    for (int b = 0; b < 4; ++b) {
        const float* up = u + ((size_t)b * 100352 + n) * 8;
        float acc = 0.f;
#pragma unroll
        for (int i = 0; i < 8; ++i) acc += up[i] * w8[i];
        uhat[(((size_t)b * 100352 + n) * 10 + j) * 16 + o] = acc;
    }
}

__global__ void zero_k(float* __restrict__ p, int nelem) {
    int t = blockIdx.x * 256 + threadIdx.x;
    if (t < nelem) p[t] = 0.f;
}

// ---------------------------------------------------------------------------
// s[b][j][o] += softmax_j(logits[b][n][:]) * u_hat[b][n][j][o]
// ---------------------------------------------------------------------------
__global__ __launch_bounds__(256) void sv_k(const float* __restrict__ logits,
                                            const float* __restrict__ uhat,
                                            float* __restrict__ s) {
    __shared__ float red[16 * 256];
    int b = blockIdx.z, j = blockIdx.y;
    int n = blockIdx.x * 256 + threadIdx.x;              // 392*256 == 100352
    const float* lp = logits + ((size_t)b * 100352 + n) * 10;
    float m = lp[0];
#pragma unroll
    for (int jj = 1; jj < 10; ++jj) m = fmaxf(m, lp[jj]);
    float den = 0.f, ej = 0.f;
#pragma unroll
    for (int jj = 0; jj < 10; ++jj) {
        float e = __expf(lp[jj] - m);
        den += e;
        if (jj == j) ej = e;
    }
    float c = ej / den;
    const float* up = uhat + (((size_t)b * 100352 + n) * 10 + j) * 16;
#pragma unroll
    for (int o = 0; o < 16; ++o) red[o * 256 + threadIdx.x] = c * up[o];
    __syncthreads();
    for (int st = 128; st > 0; st >>= 1) {
        if (threadIdx.x < st)
#pragma unroll
            for (int o = 0; o < 16; ++o)
                red[o * 256 + threadIdx.x] += red[o * 256 + threadIdx.x + st];
        __syncthreads();
    }
    if (threadIdx.x < 16)
        atomicAdd(&s[((size_t)b * 10 + j) * 16 + threadIdx.x], red[threadIdx.x * 256]);
}

__global__ void v_k(const float* __restrict__ s, float* __restrict__ v) {
    int t = threadIdx.x;
    if (t >= 40) return;
    const float* sp = s + t * 16;
    float tmp[16], n2 = 0.f;
#pragma unroll
    for (int o = 0; o < 16; ++o) { tmp[o] = sp[o]; n2 += tmp[o] * tmp[o]; }
    float sc = (n2 / (1.f + n2)) * rsqrtf(n2 + 1e-8f);
#pragma unroll
    for (int o = 0; o < 16; ++o) v[t * 16 + o] = tmp[o] * sc;
}

// logits[b][n][j] += dot(v[b][j][:], u_hat[b][n][j][:]) ; flat idx == uhat/16
__global__ void logit_k(const float* __restrict__ uhat, const float* __restrict__ v,
                        float* __restrict__ logits) {
    int t = blockIdx.x * 256 + threadIdx.x;              // 4*100352*10
    if (t >= 4014080) return;
    int j = t % 10;
    int b = t / 1003520;
    const float* up = uhat + (size_t)t * 16;
    const float* vp = v + (b * 10 + j) * 16;
    float d = 0.f;
#pragma unroll
    for (int o = 0; o < 16; ++o) d += vp[o] * up[o];
    logits[t] += d;
}

__global__ void out_k(const float* __restrict__ v, float* __restrict__ out) {
    int t = threadIdx.x;
    if (t >= 40) return;
    float n2 = 0.f;
#pragma unroll
    for (int o = 0; o < 16; ++o) { float x = v[t * 16 + o]; n2 += x * x; }
    out[t] = sqrtf(n2 + 1e-8f);
}

// ---------------------------------------------------------------------------
extern "C" void kernel_launch(void* const* d_in, const int* in_sizes, int n_in,
                              void* d_out, int out_size, void* d_ws, size_t ws_size,
                              hipStream_t stream) {
    (void)in_sizes; (void)n_in; (void)out_size; (void)ws_size;
    const float* X  = (const float*)d_in[0];
    const float* W1 = (const float*)d_in[1];
    const float* B1 = (const float*)d_in[2];
    const float* W2 = (const float*)d_in[3];
    const float* B2 = (const float*)d_in[4];
    const float* Wp = (const float*)d_in[5];
    const float* Bp = (const float*)d_in[6];
    const float* Wr = (const float*)d_in[7];
    float* out = (float*)d_out;

    char* ws = (char*)d_ws;
    size_t off = 0;
    auto take = [&](size_t bytes) -> char* {
        char* p = ws + off;
        off = (off + bytes + 255) & ~(size_t)255;
        return p;
    };
    _Float16* wt2  = (_Float16*)take((size_t)256 * 6400 * 2);        // 3.3 MB
    _Float16* wtp  = (_Float16*)take((size_t)256 * 20736 * 2);       // 10.6 MB
    _Float16* act1 = (_Float16*)take((size_t)4 * 15129 * 256 * 2);   // 31 MB
    _Float16* act2 = (_Float16*)take((size_t)4 * 14161 * 256 * 2);   // 29 MB
    float* u      = (float*)take((size_t)4 * 100352 * 8 * 4);        // 12.8 MB
    float* uhat   = (float*)take((size_t)4 * 100352 * 10 * 16 * 4);  // 257 MB
    float* logits = (float*)take((size_t)4 * 100352 * 10 * 4);       // 16 MB
    float* sbuf   = (float*)take(640 * 4);
    float* vbuf   = (float*)take(640 * 4);

    refmt_k<<<(256 * 25 * 256) / 256, 256, 0, stream>>>(W2, wt2, 25);
    refmt_k<<<(256 * 81 * 256) / 256, 256, 0, stream>>>(Wp, wtp, 81);
    conv1_k<<<(4 * 15129 * 256) / 256, 256, 0, stream>>>(X, W1, B1, act1);
    conv2_wmma<<<dim3(222, 4), 256, 0, stream>>>(act1, wt2, B2, act2);
    convp_wmma<<<dim3(49, 4), 256, 0, stream>>>(act2, wtp, Bp, u);
    squash_k<<<(401408 + 255) / 256, 256, 0, stream>>>(u);
    uhat_k<<<16056320 / 256, 256, 0, stream>>>(u, Wr, uhat);
    zero_k<<<(4014080 + 255) / 256, 256, 0, stream>>>(logits, 4014080);
    for (int it = 0; it < 3; ++it) {
        zero_k<<<3, 256, 0, stream>>>(sbuf, 640);
        sv_k<<<dim3(392, 10, 4), 256, 0, stream>>>(logits, uhat, sbuf);
        v_k<<<1, 64, 0, stream>>>(sbuf, vbuf);
        if (it < 2)
            logit_k<<<4014080 / 256, 256, 0, stream>>>(uhat, vbuf, logits);
    }
    out_k<<<1, 64, 0, stream>>>(vbuf, out);
}
